// SPINEModel_26903675142682
// MI455X (gfx1250) — compile-verified, matrix operand order checked by reading, not run on hardware
//
#include <hip/hip_runtime.h>
#include <hip/hip_bf16.h>
#include <math.h>

typedef _Float16 v16h __attribute__((ext_vector_type(16)));
typedef _Float16 v8h  __attribute__((ext_vector_type(8)));
typedef float    v8f  __attribute__((ext_vector_type(8)));

#define B_ROWS      8192
#define INP         300
#define INP_PAD     320
#define HD          1000
#define HD_PAD      1024
#define W1_ROWS_PAD 1008   // 63 tiles of 16
#define W2_ROWS_PAD 304    // 19 tiles of 16
#define KTOP        10

// --- WMMA fragment load: row-major f16 row pointer `p` at K-offset, lane-half `hi`.
// ISA layout (16-bit A/B 16x32): lanes 0-15 hold K=[0..7]+[16..23], lanes 16-31 hold K=[8..15]+[24..31].
__device__ __forceinline__ v16h load_frag16(const _Float16* p, int hi) {
  v8h lo = *(const v8h*)(p + hi * 8);        // K = hi*8 .. hi*8+7
  v8h up = *(const v8h*)(p + 16 + hi * 8);   // K = 16+hi*8 .. +7
  return __builtin_shufflevector(lo, up, 0,1,2,3,4,5,6,7,8,9,10,11,12,13,14,15);
}

// ---------------- conversion / padding ----------------
__global__ void pad_to_f16(const float* __restrict__ src, _Float16* __restrict__ dst,
                           int rows, int cols, int colsPad, long total) {
  long i = (long)blockIdx.x * blockDim.x + threadIdx.x;
  if (i >= total) return;
  int r = (int)(i / colsPad), c = (int)(i % colsPad);
  float v = (r < rows && c < cols) ? src[(long)r * cols + c] : 0.f;
  dst[i] = (_Float16)v;
}

// ---------------- row-normalize (cosine prep), f16 output ----------------
__global__ void row_norm_f16(const float* __restrict__ src, _Float16* __restrict__ dstN,
                             int cols, int colsPad) {
  int row = blockIdx.x, tid = threadIdx.x;
  __shared__ float red[128];
  const float* sr = src + (long)row * cols;
  float ss = 0.f;
  for (int c = tid; c < cols; c += blockDim.x) { float v = sr[c]; ss += v * v; }
  red[tid] = ss; __syncthreads();
  for (int s = blockDim.x >> 1; s > 0; s >>= 1) { if (tid < s) red[tid] += red[tid + s]; __syncthreads(); }
  float inv = 1.f / fmaxf(sqrtf(red[0]), 1e-6f);
  for (int c = tid; c < colsPad; c += blockDim.x) {
    float v = (c < cols) ? sr[c] * inv : 0.f;
    dstN[(long)row * colsPad + c] = (_Float16)v;
  }
}

// h -> f16 copy (padded) + f16 normalized copy, one pass
__global__ void h_prep(const float* __restrict__ h, _Float16* __restrict__ h16,
                       _Float16* __restrict__ h16n) {
  int row = blockIdx.x, tid = threadIdx.x;
  __shared__ float red[256];
  const float* hr = h + (long)row * HD;
  float ss = 0.f;
  for (int c = tid; c < HD; c += 256) { float v = hr[c]; ss += v * v; }
  red[tid] = ss; __syncthreads();
  for (int s = 128; s > 0; s >>= 1) { if (tid < s) red[tid] += red[tid + s]; __syncthreads(); }
  float inv = 1.f / fmaxf(sqrtf(red[0]), 1e-6f);
  for (int c = tid; c < HD_PAD; c += 256) {
    float v = (c < HD) ? hr[c] : 0.f;
    h16 [(long)row * HD_PAD + c] = (_Float16)v;
    h16n[(long)row * HD_PAD + c] = (_Float16)(v * inv);
  }
}

// ---------------- WMMA GEMM: out[M x nValid] = A[M x kDim] * B[N x kDim]^T + bias ----------------
__global__ __launch_bounds__(256)
void wmma_gemm_bias(const _Float16* __restrict__ A, const _Float16* __restrict__ Bm,
                    const float* __restrict__ bias, float* __restrict__ out,
                    int colTiles, int kDim, int nValid, int ldOut, int doClamp) {
  int tid = threadIdx.x;
  int wid = tid >> 5, lane = tid & 31;
  int tile = blockIdx.x * 8 + wid;              // grid sized so tile always valid
  int rowTile = tile / colTiles, colTile = tile % colTiles;
  int r = lane & 15, hi = lane >> 4;
  const _Float16* arow = A  + (size_t)(rowTile * 16 + r) * kDim;
  const _Float16* brow = Bm + (size_t)(colTile * 16 + r) * kDim;
  v8f acc = {};
  for (int kk = 0; kk < kDim; kk += 32) {
    v16h a = load_frag16(arow + kk, hi);
    v16h b = load_frag16(brow + kk, hi);
    acc = __builtin_amdgcn_wmma_f32_16x16x32_f16(false, a, false, b, (short)0, acc, false, false);
  }
  int col = colTile * 16 + r;                   // C layout: lane owns column N=r, rows hi*8+v
  if (col < nValid) {
    float bv = bias[col];
#pragma unroll
    for (int v = 0; v < 8; v++) {
      int row = rowTile * 16 + hi * 8 + v;
      float x = acc[v] + bv;
      if (doClamp) x = fminf(fmaxf(x, 0.f), 1.f);
      out[(size_t)row * ldOut + col] = x;
    }
  }
}

// ---------------- fused similarity (xn * xn^T) + streaming per-row top-10 ----------------
template <int KDIM>
__global__ __launch_bounds__(128)
void wmma_topk(const _Float16* __restrict__ norm, float* __restrict__ topOut) {
  __shared__ __align__(16) _Float16 As[16 * KDIM];  // this block's 16 rows (A operand, reused all sweep)
  __shared__ float Cs[16][68];                      // 16x64 staged tile (+pad)
  __shared__ float tk[16][KTOP];                    // sorted-descending top-10 per row
  int tid = threadIdx.x, wid = tid >> 5, lane = tid & 31;
  int rowBase = blockIdx.x * 16;

  { // stage A tile into LDS via gfx1250 async data mover (cache -> LDS, no VGPR round trip).
    // 16*KDIM/8 sixteen-byte chunks; trip count is a multiple of 128 so EXEC is all-ones
    // for every issue (each lane copies 16B global -> LDS).
    const _Float16* gsrc = norm + (size_t)rowBase * KDIM;
    const int n16 = 16 * KDIM / 8;
    for (int i = tid; i < n16; i += 128) {
      unsigned ldsOff = (unsigned)(unsigned long long)(uintptr_t)&As[i * 8]; // LDS aperture: addr[31:0]
      unsigned long long gaddr = (unsigned long long)(uintptr_t)(gsrc + i * 8);
      asm volatile("global_load_async_to_lds_b128 %0, %1, off"
                   :: "v"(ldsOff), "v"(gaddr) : "memory");
    }
    asm volatile("s_wait_asynccnt 0x0" ::: "memory");
  }
  if (tid < 16)
    for (int j = 0; j < KTOP; j++) tk[tid][j] = -1e30f;
  __syncthreads();

  int r = lane & 15, hi = lane >> 4;
  const _Float16* arow = As + r * KDIM;
  for (int j0 = 0; j0 < B_ROWS; j0 += 64) {        // 4 waves sweep 64 columns / iter
    const _Float16* brow = norm + (size_t)(j0 + wid * 16 + r) * KDIM;
    if (j0 + 64 < B_ROWS) {                        // warm next slab (global_prefetch_b8)
      __builtin_prefetch(norm + (size_t)(j0 + 64 + wid * 16 + r) * KDIM, 0, 3);
    }
    v8f acc = {};
    for (int kk = 0; kk < KDIM; kk += 32) {
      v16h a = load_frag16(arow + kk, hi);         // LDS
      v16h b = load_frag16(brow + kk, hi);         // global (L2-resident)
      acc = __builtin_amdgcn_wmma_f32_16x16x32_f16(false, a, false, b, (short)0, acc, false, false);
    }
#pragma unroll
    for (int v = 0; v < 8; v++) Cs[hi * 8 + v][wid * 16 + r] = acc[v];
    __syncthreads();
    if (tid < 16) {                                // one thread per row maintains sorted top-10
      int grow = rowBase + tid;
      float* list = tk[tid];
      for (int c = 0; c < 64; c++) {
        float val = Cs[tid][c];
        if (j0 + c == grow) val = -10.f;           // reference: diagonal = -10
        if (val > list[KTOP - 1]) {
          int p = KTOP - 1;
          while (p > 0 && list[p - 1] < val) { list[p] = list[p - 1]; --p; }
          list[p] = val;
        }
      }
    }
    __syncthreads();
  }
  if (tid < 16 * KTOP)
    topOut[(size_t)rowBase * KTOP + tid] = tk[tid / KTOP][tid % KTOP];
}

// ---------------- reductions ----------------
__global__ void col_stats(const float* __restrict__ h, float* __restrict__ colsum,
                          float* __restrict__ pslAcc) {
  int c = blockIdx.x, tid = threadIdx.x;
  __shared__ float r0[256], r1[256];
  float s = 0.f, p = 0.f;
  for (int r = tid; r < B_ROWS; r += 256) {
    float v = h[(size_t)r * HD + c];
    s += v; p += v * (1.f - v);
  }
  r0[tid] = s; r1[tid] = p; __syncthreads();
  for (int st = 128; st > 0; st >>= 1) {
    if (tid < st) { r0[tid] += r0[tid + st]; r1[tid] += r1[tid + st]; }
    __syncthreads();
  }
  if (tid == 0) { colsum[c] = r0[0]; atomicAdd(pslAcc, r1[0]); }
}

__global__ void sq_err(const float* __restrict__ o, const float* __restrict__ y,
                       float* __restrict__ acc, long n) {
  long i0 = (long)blockIdx.x * blockDim.x + threadIdx.x;
  long stride = (long)gridDim.x * blockDim.x;
  float s = 0.f;
  for (long i = i0; i < n; i += stride) { float d = o[i] - y[i]; s += d * d; }
  __shared__ float red[256];
  red[threadIdx.x] = s; __syncthreads();
  for (int st = 128; st > 0; st >>= 1) { if (threadIdx.x < st) red[threadIdx.x] += red[threadIdx.x + st]; __syncthreads(); }
  if (threadIdx.x == 0) atomicAdd(acc, red[0]);
}

__global__ void zero_accs(float* p) { if (threadIdx.x < 2) p[threadIdx.x] = 0.f; }

__global__ void finalize(const float* __restrict__ colsum, const float* __restrict__ ty,
                         const float* __restrict__ th, const float* __restrict__ reconAcc,
                         const float* __restrict__ pslAcc, float* __restrict__ scal) {
  __shared__ float red[256];
  int tid = threadIdx.x;
  float a = 0.f;
  for (int c = tid; c < HD; c += 256) {
    float t = fmaxf(colsum[c] * (1.f / B_ROWS) - 0.15f, 0.f);  // rho* = 1-0.85
    a += t * t;
  }
  red[tid] = a; __syncthreads();
  for (int s = 128; s > 0; s >>= 1) { if (tid < s) red[tid] += red[tid + s]; __syncthreads(); }
  float asl = (tid == 0) ? red[0] / (float)HD : 0.f;
  __syncthreads();
  float l = 0.f;
  for (int i = tid; i < B_ROWS * KTOP; i += 256) l += fabsf(ty[i] - th[i]);
  red[tid] = l; __syncthreads();
  for (int s = 128; s > 0; s >>= 1) { if (tid < s) red[tid] += red[tid + s]; __syncthreads(); }
  if (tid == 0) {
    float local = red[0] / (float)(B_ROWS * KTOP);
    float recon = reconAcc[0] / (float)((long)B_ROWS * INP);
    float psl   = pslAcc[0] / (float)((long)B_ROWS * HD);
    scal[0] = recon + psl + asl + local;
    scal[1] = recon; scal[2] = psl; scal[3] = asl; scal[4] = local;
  }
}

// ---------------- launch ----------------
extern "C" void kernel_launch(void* const* d_in, const int* in_sizes, int n_in,
                              void* d_out, int out_size, void* d_ws, size_t ws_size,
                              hipStream_t stream) {
  (void)in_sizes; (void)n_in; (void)out_size; (void)ws_size;
  const float* x  = (const float*)d_in[0];
  const float* y  = (const float*)d_in[1];
  const float* W1 = (const float*)d_in[2];
  const float* b1 = (const float*)d_in[3];
  const float* W2 = (const float*)d_in[4];
  const float* b2 = (const float*)d_in[5];

  float* out  = (float*)d_out;                   // [8192*300]
  float* h    = out + (size_t)B_ROWS * INP;      // [8192*1000]
  float* scal = h + (size_t)B_ROWS * HD;         // 5 scalars

  char* w = (char*)d_ws;
  auto carve = [&](size_t bytes) -> void* {
    void* p = (void*)w; w += (bytes + 255) & ~(size_t)255; return p;
  };
  _Float16* x16  = (_Float16*)carve((size_t)B_ROWS * INP_PAD * 2);
  _Float16* w116 = (_Float16*)carve((size_t)W1_ROWS_PAD * INP_PAD * 2);
  _Float16* w216 = (_Float16*)carve((size_t)W2_ROWS_PAD * HD_PAD * 2);
  _Float16* y16n = (_Float16*)carve((size_t)B_ROWS * INP_PAD * 2);
  _Float16* h16  = (_Float16*)carve((size_t)B_ROWS * HD_PAD * 2);
  _Float16* h16n = (_Float16*)carve((size_t)B_ROWS * HD_PAD * 2);
  float* topY   = (float*)carve((size_t)B_ROWS * KTOP * 4);
  float* topH   = (float*)carve((size_t)B_ROWS * KTOP * 4);
  float* colsum = (float*)carve((size_t)HD * 4);
  float* accs   = (float*)carve(2 * 4);          // [recon, psl]

  zero_accs<<<1, 32, 0, stream>>>(accs);
  { long n = (long)B_ROWS * INP_PAD;      pad_to_f16<<<(int)((n + 255) / 256), 256, 0, stream>>>(x,  x16,  B_ROWS, INP, INP_PAD, n); }
  { long n = (long)W1_ROWS_PAD * INP_PAD; pad_to_f16<<<(int)((n + 255) / 256), 256, 0, stream>>>(W1, w116, HD,     INP, INP_PAD, n); }
  { long n = (long)W2_ROWS_PAD * HD_PAD;  pad_to_f16<<<(int)((n + 255) / 256), 256, 0, stream>>>(W2, w216, INP,    HD,  HD_PAD,  n); }
  row_norm_f16<<<B_ROWS, 128, 0, stream>>>(y, y16n, INP, INP_PAD);

  // h = clamp(x @ W1^T + b1, 0, 1): 512 row-tiles x 63 col-tiles = 32256 wave-tiles / 8 per block
  wmma_gemm_bias<<<(B_ROWS / 16) * 63 / 8, 256, 0, stream>>>(x16, w116, b1, h, 63, INP_PAD, HD, HD, 1);
  h_prep<<<B_ROWS, 256, 0, stream>>>(h, h16, h16n);
  col_stats<<<HD, 256, 0, stream>>>(h, colsum, accs + 1);

  // out = h @ W2^T + b2: 512 x 19 = 9728 wave-tiles / 8 per block
  wmma_gemm_bias<<<(B_ROWS / 16) * 19 / 8, 256, 0, stream>>>(h16, w216, b2, out, 19, HD_PAD, INP, INP, 0);
  sq_err<<<2048, 256, 0, stream>>>(out, y, accs + 0, (long)B_ROWS * INP);

  wmma_topk<INP_PAD><<<B_ROWS / 16, 128, 0, stream>>>(y16n, topY);
  wmma_topk<HD_PAD><<<B_ROWS / 16, 128, 0, stream>>>(h16n, topH);

  finalize<<<1, 256, 0, stream>>>(colsum, topY, topH, accs + 0, accs + 1, scal);
}
